// RNN_s_76381698392696
// MI455X (gfx1250) — matrix operand order repaired and checked by reading, not compiled, not run on hardware
//
#include <hip/hip_runtime.h>
#include <math.h>

#define HID   100
#define SEQ   1000
#define IN_D  8
#define NPAD  128     // padded neuron / K dimension
#define BTILE 16      // batch rows per workgroup

typedef __attribute__((ext_vector_type(16))) __bf16 v16bf;
typedef __attribute__((ext_vector_type(8)))  __bf16 v8bf;
typedef __attribute__((ext_vector_type(8)))  float  v8f;

struct LayerPtrs {
  const float *Win1, *b1, *Wrec1, *taum1, *tauadp1;
  const float *Win2, *b2, *Wrec2, *taum2, *tauadp2;
};

struct Params {
  const float* inp;       // [128][1000][8]
  LayerPtrs chain[2];     // 0 = fw (fw1,fw2), 1 = bw (bw1,bw2)
  const float* roW;       // [200][2] row-major
  const float* rotaum;    // [2]
  float* partial;         // d_ws: [2 chains][128 rows][2 outs]
};

// ---- helpers -------------------------------------------------------------

__device__ inline v16bf zero16() {
  v16bf z;
  #pragma unroll
  for (int j = 0; j < 16; ++j) z[j] = (__bf16)0.f;
  return z;
}

__device__ inline v8f splat8(float x) {
  v8f v;
  #pragma unroll
  for (int r = 0; r < 8; ++r) v[r] = x;
  return v;
}

// B-fragment (32x16, K x N) for one wave's 16-column tile.
// Layout: lanes 0-15 hold K=kb..kb+15 of column (c0+lane), lanes 16-31 hold
// K=kb+16..kb+31 of column (c0+lane-16); 16 bf16 per lane, K ascending.
__device__ inline v16bf load_bfrag(const float* W, int Kdim, int kb, int c0, int lane) {
  const int n    = lane & 15;
  const int col  = c0 + n;
  const int kbase = kb + ((lane >> 4) << 4);
  v16bf f;
  #pragma unroll
  for (int j = 0; j < 16; ++j) {
    const int k = kbase + j;
    float v = (k < Kdim && col < HID) ? W[k * HID + col] : 0.f;
    f[j] = (__bf16)v;
  }
  return f;
}

// A-fragment (16x32, M x K) from an LDS [16][NPAD] bf16 tile.
// Layout: lane<16 -> row=lane, K = kb..kb+15;
// lane>=16 -> row=lane-16, K = kb+16..kb+31 (matches load_bfrag).
__device__ inline v16bf load_afrag(const __bf16* buf, int kb, int lane) {
  const int m   = lane & 15;
  const int klo = kb + ((lane >> 4) << 4);
  const v8bf lo = *(const v8bf*)(buf + m * NPAD + klo);        // ds_load_b128
  const v8bf hi = *(const v8bf*)(buf + m * NPAD + klo + 8);    // ds_load_b128
  return __builtin_shufflevector(lo, hi, 0,1,2,3,4,5,6,7,8,9,10,11,12,13,14,15);
}

// Adaptive-LIF elementwise update for 8 (row) elements of one neuron column.
__device__ inline v8f lif_update(v8f& mem, v8f& bb, v8f& spk, v8f cur,
                                 float alpha, float ro, float valid) {
  v8f snew;
  #pragma unroll
  for (int r = 0; r < 8; ++r) {
    float b2  = ro * bb[r] + (1.f - ro) * spk[r];
    float thr = 0.01f + 1.8f * b2;                       // B_J0 + BETA*b
    float m2  = alpha * mem[r] + (1.f - alpha) * cur[r] - thr * spk[r];
    float s   = (m2 > thr) ? valid : 0.f;                // spike_fn, pad-masked
    bb[r] = b2; mem[r] = m2; snew[r] = s;
  }
  spk = snew;
  return snew;
}

#define WMMA_BF16(A, B, C) \
  __builtin_amdgcn_wmma_f32_16x16x32_bf16(false, (A), false, (B), (short)0, (C), false, false)

// ---- main persistent chain kernel ---------------------------------------

__global__ __launch_bounds__(256, 1)
void srnn_chain_kernel(Params p) {
  const int tid     = threadIdx.x;
  const int lane    = tid & 31;
  const int wave    = tid >> 5;           // 8 waves: one 16-col N-tile each
  const int chain   = blockIdx.x & 1;     // 0 = fw, 1 = bw
  const int rowbase = (blockIdx.x >> 1) * BTILE;
  const LayerPtrs sp = p.chain[chain];

  __shared__ __align__(16) __bf16 spk1buf[2][BTILE][NPAD];
  __shared__ __align__(16) __bf16 spk2buf[2][BTILE][NPAD];
  __shared__ float accbuf[BTILE][2];

  for (int i = tid; i < 2 * BTILE * NPAD; i += 256) {
    ((__bf16*)spk1buf)[i] = (__bf16)0.f;
    ((__bf16*)spk2buf)[i] = (__bf16)0.f;
  }
  if (tid < BTILE * 2) ((float*)accbuf)[tid] = 0.f;

  const int   c0    = wave * 16;
  const int   n     = lane & 15;
  const int   col   = c0 + n;
  const bool  vcol  = col < HID;
  const float valid = vcol ? 1.f : 0.f;
  const int   mbase = (lane >> 4) * 8;    // rows 0-7 / 8-15 of the D tile

  float alpha1 = 0.f, ro1 = 0.f, bias1 = 0.f;
  float alpha2 = 0.f, ro2 = 0.f, bias2 = 0.f;
  float w0 = 0.f, w1 = 0.f;
  if (vcol) {
    alpha1 = __expf(-1.f / sp.taum1[col]);
    ro1    = __expf(-1.f / sp.tauadp1[col]);
    bias1  = sp.b1[col];
    alpha2 = __expf(-1.f / sp.taum2[col]);
    ro2    = __expf(-1.f / sp.tauadp2[col]);
    bias2  = sp.b2[col];
    w0 = p.roW[(chain * HID + col) * 2 + 0];   // fw uses rows 0..99, bw 100..199
    w1 = p.roW[(chain * HID + col) * 2 + 1];
  }
  const float aro0 = __expf(-1.f / p.rotaum[0]);
  const float aro1 = __expf(-1.f / p.rotaum[1]);
  const float k0 = (1.f - aro0) * w0;
  const float k1 = (1.f - aro1) * w1;

  // Weights resident in VGPRs for the whole scan.
  v16bf win1 = load_bfrag(sp.Win1, IN_D, 0, c0, lane);
  v16bf wrec1[4], win2[4], wrec2[4];
  #pragma unroll
  for (int kb = 0; kb < 4; ++kb) {
    wrec1[kb] = load_bfrag(sp.Wrec1, HID, kb * 32, c0, lane);
    win2[kb]  = load_bfrag(sp.Win2,  HID, kb * 32, c0, lane);
    wrec2[kb] = load_bfrag(sp.Wrec2, HID, kb * 32, c0, lane);
  }

  v8f mem1 = splat8(0.f), bb1 = splat8(0.01f), spk1 = splat8(0.f);
  v8f mem2 = splat8(0.f), bb2 = splat8(0.01f), spk2 = splat8(0.f);
  v8f acc0v = splat8(0.f), acc1v = splat8(0.f);

  __syncthreads();

  #pragma unroll 1
  for (int t = 0; t < SEQ; ++t) {
    const int cur = t & 1, prev = cur ^ 1;
    const int tt  = chain ? (SEQ - 1 - t) : t;   // bw chain reads reversed input

    // ---------------- layer 1 ----------------
    v16bf a_in = zero16();
    if (lane < 16) {
      const float* xr = p.inp + ((size_t)(rowbase + n) * SEQ + tt) * IN_D;
      #pragma unroll
      for (int j = 0; j < IN_D; ++j) a_in[j] = (__bf16)xr[j];
    }
    v8f c = splat8(bias1);
    c = WMMA_BF16(a_in, win1, c);
    #pragma unroll
    for (int kb = 0; kb < 4; ++kb) {
      v16bf a = load_afrag(&spk1buf[prev][0][0], kb * 32, lane);
      c = WMMA_BF16(a, wrec1[kb], c);
    }
    v8f s1 = lif_update(mem1, bb1, spk1, c, alpha1, ro1, valid);
    #pragma unroll
    for (int r = 0; r < 8; ++r) spk1buf[cur][mbase + r][col] = (__bf16)s1[r];
    __syncthreads();

    // ---------------- layer 2 ----------------
    v8f c2 = splat8(bias2);
    #pragma unroll
    for (int kb = 0; kb < 4; ++kb) {
      v16bf a = load_afrag(&spk1buf[cur][0][0], kb * 32, lane);
      c2 = WMMA_BF16(a, win2[kb], c2);
    }
    #pragma unroll
    for (int kb = 0; kb < 4; ++kb) {
      v16bf a = load_afrag(&spk2buf[prev][0][0], kb * 32, lane);
      c2 = WMMA_BF16(a, wrec2[kb], c2);
    }
    v8f s2 = lif_update(mem2, bb2, spk2, c2, alpha2, ro2, valid);
    #pragma unroll
    for (int r = 0; r < 8; ++r) spk2buf[cur][mbase + r][col] = (__bf16)s2[r];

    // Readout folded into the scan (linear recurrence, per-lane partial).
    #pragma unroll
    for (int r = 0; r < 8; ++r) {
      acc0v[r] = aro0 * acc0v[r] + k0 * s2[r];
      acc1v[r] = aro1 * acc1v[r] + k1 * s2[r];
    }
    __syncthreads();
  }

  // Reduce per-lane readout partials across the 128 neuron columns.
  #pragma unroll
  for (int r = 0; r < 8; ++r) {
    atomicAdd(&accbuf[mbase + r][0], acc0v[r]);
    atomicAdd(&accbuf[mbase + r][1], acc1v[r]);
  }
  __syncthreads();
  if (tid < BTILE * 2) {
    const int row = tid >> 1, o = tid & 1;
    p.partial[chain * 256 + (rowbase + row) * 2 + o] = accbuf[row][o];
  }
}

// ---- epilogue: combine fw/bw partials, bias term, log_softmax -----------

__global__ void srnn_finalize_kernel(const float* partial, const float* ro_b,
                                     const float* ro_taum, float* out) {
  const int b = threadIdx.x;
  if (b >= 128) return;
  float v[2];
  #pragma unroll
  for (int o = 0; o < 2; ++o) {
    float a    = __expf(-1.f / ro_taum[o]);
    float bias = ro_b[o] * (1.f - __powf(a, (float)SEQ));   // closed-form bias term
    v[o] = partial[b * 2 + o] + partial[256 + b * 2 + o] + bias;
  }
  float mx  = fmaxf(v[0], v[1]);
  float lse = mx + __logf(__expf(v[0] - mx) + __expf(v[1] - mx));
  out[b * 2 + 0] = v[0] - lse;
  out[b * 2 + 1] = v[1] - lse;
}

// ---- launch --------------------------------------------------------------

extern "C" void kernel_launch(void* const* d_in, const int* in_sizes, int n_in,
                              void* d_out, int out_size, void* d_ws, size_t ws_size,
                              hipStream_t stream) {
  (void)in_sizes; (void)n_in; (void)out_size; (void)ws_size;
  Params p;
  p.inp = (const float*)d_in[0];
  const int base1[2] = {1, 6};    // fw1, bw1
  const int base2[2] = {11, 16};  // fw2, bw2
  for (int c = 0; c < 2; ++c) {
    p.chain[c].Win1    = (const float*)d_in[base1[c] + 0];
    p.chain[c].b1      = (const float*)d_in[base1[c] + 1];
    p.chain[c].Wrec1   = (const float*)d_in[base1[c] + 2];
    p.chain[c].taum1   = (const float*)d_in[base1[c] + 3];
    p.chain[c].tauadp1 = (const float*)d_in[base1[c] + 4];
    p.chain[c].Win2    = (const float*)d_in[base2[c] + 0];
    p.chain[c].b2      = (const float*)d_in[base2[c] + 1];
    p.chain[c].Wrec2   = (const float*)d_in[base2[c] + 2];
    p.chain[c].taum2   = (const float*)d_in[base2[c] + 3];
    p.chain[c].tauadp2 = (const float*)d_in[base2[c] + 4];
  }
  p.roW     = (const float*)d_in[21];
  p.rotaum  = (const float*)d_in[23];
  p.partial = (float*)d_ws;

  // 16 WGs: {fw,bw} x 8 batch tiles of 16 rows, each persistent over SEQ steps.
  hipLaunchKernelGGL(srnn_chain_kernel, dim3(16), dim3(256), 0, stream, p);
  hipLaunchKernelGGL(srnn_finalize_kernel, dim3(1), dim3(128), 0, stream,
                     (const float*)d_ws, (const float*)d_in[22],
                     (const float*)d_in[23], (float*)d_out);
}
